// STGNNAdapter_82360292868128
// MI455X (gfx1250) — matrix operand order repaired and checked by reading, not compiled, not run on hardware
//
#include <hip/hip_runtime.h>
#include <hip/hip_bf16.h>

typedef __attribute__((ext_vector_type(16))) _Float16 v16h;
typedef __attribute__((ext_vector_type(8)))  _Float16 v8h;
typedef __attribute__((ext_vector_type(8)))  float    v8f;

#define EPSV 1e-5f

// ---------------------------------------------------------------------------
// WMMA fragment index maps (CDNA5 ISA §7.12.2, wave32)
// A (16x32 f16): lane l -> m=l&15 ; slot s -> k = hi*16+s (this code's order, not the ISA table)
//                => one run of 16 consecutive K: [hi*16, hi*16+15]
// B (32x16 f16): lane l -> n=l&15 ; slot s -> k = hi*16+s (this code's order, not the ISA table)
//                => one run of 16 consecutive K: [hi*16, hi*16+15]
// D (16x16 f32): lane l -> n=l&15 ; vgpr v -> m = v + 8*hi
// ---------------------------------------------------------------------------
__device__ __forceinline__ int a_slot_k(int s, int hi) {
  return hi * 16 + s;
}

// ---------------------------------------------------------------------------
// Kernel 0: repack weights to f16 (zero-padded so gathers are branchless).
//  w2g  [5][32][96] : row r=out-ch-in-group (pad 19..31 = 0), k=ci*5+t (k=95 pad)
//  w4c  [32][160]   : flat convert of w4 (rows = 320B, 32B aligned)
//  fc1T [32][2048]  : flat convert of fc1_w (rows contiguous in K)
// ---------------------------------------------------------------------------
__global__ __launch_bounds__(256) void k_prep(const float* __restrict__ w2,
                                              const float* __restrict__ w4,
                                              const float* __restrict__ fc1w,
                                              _Float16* __restrict__ w2g,
                                              _Float16* __restrict__ w4c,
                                              _Float16* __restrict__ fc1T) {
  int i = blockIdx.x * 256 + threadIdx.x;
  if (i < 5 * 32 * 96) {
    int g = i / (32 * 96), r = (i / 96) % 32, k = i % 96;
    float v = 0.f;
    if (r < 19 && k < 95) v = w2[((g * 19 + r) * 19 + (k / 5)) * 5 + (k % 5)];
    w2g[i] = (_Float16)v;
  }
  if (i < 32 * 160) w4c[i] = (_Float16)w4[i];
  if (i < 32 * 2048) fc1T[i] = (_Float16)fc1w[i];
}

// ---------------------------------------------------------------------------
// Kernel 1: fused transpose + conv1(dw,k7)+BN1+ReLU (LDS)
//           + conv2(groups=5,k5) via WMMA f16 + BN2+ReLU + avgpool60.
// Grid: (B=64, tiles=64), block 256 (8 waves). Tile = 240 positions.
// sH1 padded to [5][20][248] (20th row of each group zeroed) so the K=96
// implicit-GEMM gather needs no predication.
// ---------------------------------------------------------------------------
__global__ __launch_bounds__(256) void k_front(
    const float* __restrict__ x,
    const float* __restrict__ w1, const float* __restrict__ b1,
    const float* __restrict__ g1, const float* __restrict__ bb1,
    const float* __restrict__ m1, const float* __restrict__ v1,
    const _Float16* __restrict__ w2g, const float* __restrict__ b2,
    const float* __restrict__ g2, const float* __restrict__ bb2,
    const float* __restrict__ m2, const float* __restrict__ v2,
    float* __restrict__ pooled) {
  __shared__ _Float16 sIn[19][252];      // 250 used: pos t0-5 .. t0+244
  __shared__ _Float16 sH1[100][248];     // [5 groups][20 rows][248]
  __shared__ float    sPool[95][4];

  const int b    = blockIdx.x;
  const int tile = blockIdx.y;
  const int t0   = tile * 240;
  const int tid  = threadIdx.x;

  for (int i = tid; i < 95 * 4; i += 256) (&sPool[0][0])[i] = 0.f;

  // -- input tile (branchless boundary: clamp + mask) ------------------------
  for (int i = tid; i < 19 * 250; i += 256) {
    int c = i / 250, j = i % 250;
    int p = t0 - 5 + j;
    int pc = min(max(p, 0), 15359);
    int t = pc >> 9, l = pc & 511;
    float val = x[(((size_t)b * 30 + t) * 19 + c) * 512 + l];
    sIn[c][j] = (_Float16)((p == pc) ? val : 0.f);
  }
  // -- zero the pad row of each group (row 19) -------------------------------
  for (int i = tid; i < 5 * 248; i += 256)
    sH1[(i / 248) * 20 + 19][i % 248] = (_Float16)0.f;
  __syncthreads();

  // -- conv1 + BN1 + ReLU -> sH1 (channel ch -> row ch + ch/19) --------------
  for (int i = tid; i < 95 * 244; i += 256) {
    int ch = i / 244, j = i % 244;
    int c = ch / 5;
    float acc = b1[ch];
#pragma unroll
    for (int k = 0; k < 7; k++) acc += (float)sIn[c][j + k] * w1[ch * 7 + k];
    float s = g1[ch] * rsqrtf(v1[ch] + EPSV);
    float val = (acc - m1[ch]) * s + bb1[ch];
    sH1[ch + ch / 19][j] = (_Float16)fmaxf(val, 0.f);
  }
  __syncthreads();

  // -- conv2 implicit GEMM: jobs = (group 5) x (mtile 15); A shared by 2 nt --
  const int wave = tid >> 5, lane = tid & 31;
  const int mL = lane & 15, hi = lane >> 4;
  for (int job = wave; job < 75; job += 8) {
    int g = job / 15, mt = job % 15;
    int j0 = mt * 16;
    v8f acc0 = {}, acc1 = {};
#pragma unroll
    for (int kc = 0; kc < 3; kc++) {
      v16h A;
#pragma unroll
      for (int s = 0; s < 16; s++) {
        int k = kc * 32 + a_slot_k(s, hi);   // < 96; k=95 -> pad row (zero)
        int ci = k / 5, tp = k % 5;
        A[s] = sH1[g * 20 + ci][j0 + mL + tp];
      }
      // B: one contiguous 16-half (32B-aligned) run per lane
      const v16h B0 = *(const v16h*)(w2g + ((size_t)(g * 32 + mL) * 96 + kc * 32 + hi * 16));
      const v16h B1 = *(const v16h*)(w2g + ((size_t)(g * 32 + 16 + mL) * 96 + kc * 32 + hi * 16));
      acc0 = __builtin_amdgcn_wmma_f32_16x16x32_f16(false, A, false, B0,
                                                    (short)0, acc0, false, false);
      acc1 = __builtin_amdgcn_wmma_f32_16x16x32_f16(false, A, false, B1,
                                                    (short)0, acc1, false, false);
    }
    // BN2 + ReLU + pool60 (lane owns channel col; 8 position rows)
#pragma unroll
    for (int nt = 0; nt < 2; nt++) {
      const v8f& acc = nt ? acc1 : acc0;
      int nch = nt * 16 + mL;
      if (nch < 19) {
        int o = g * 19 + nch;
        float s2 = g2[o] * rsqrtf(v2[o] + EPSV);
        float sh = bb2[o] + (b2[o] - m2[o]) * s2;
#pragma unroll
        for (int v = 0; v < 8; v++) {
          int j = j0 + v + 8 * hi;              // 0..239
          float val = fmaxf(acc[v] * s2 + sh, 0.f);
          atomicAdd(&sPool[o][j / 60], val);    // ds_add_f32
        }
      }
    }
  }
  __syncthreads();

  // -- write pooled mean in (B,5,19,256) layout ------------------------------
  for (int i = tid; i < 95 * 4; i += 256) {
    int o = i >> 2, bin = i & 3;
    pooled[(((size_t)b * 5 + (o % 5)) * 19 + (o / 5)) * 256 + tile * 4 + bin] =
        sPool[o][bin] * (1.f / 60.f);
  }
}

// ---------------------------------------------------------------------------
// Kernel 2: conv3 (5 -> 32, k(1,7), pad 3) + BN3 + ReLU (scalar, K=35).
// Grid: B*19 blocks, 128 threads. Out: h3 f16 [(b*19+c)][32][256].
// ---------------------------------------------------------------------------
__global__ __launch_bounds__(128) void k_conv3(
    const float* __restrict__ pooled, const float* __restrict__ w3,
    const float* __restrict__ b3, const float* __restrict__ g3,
    const float* __restrict__ bb3, const float* __restrict__ m3,
    const float* __restrict__ v3, _Float16* __restrict__ h3) {
  __shared__ float sI[5][264];   // 262 used, w = -3..258
  int bc = blockIdx.x;
  int b = bc / 19, c = bc % 19;
  int tid = threadIdx.x;
  for (int i = tid; i < 5 * 262; i += 128) {
    int nb = i / 262, j = i % 262;
    int w = j - 3;
    int wc = min(max(w, 0), 255);
    float val = pooled[(((size_t)b * 5 + nb) * 19 + c) * 256 + wc];
    sI[nb][j] = (w == wc) ? val : 0.f;
  }
  __syncthreads();
  for (int i = tid; i < 32 * 256; i += 128) {
    int o = i >> 8, w = i & 255;
    float acc = b3[o];
    for (int nb = 0; nb < 5; nb++) {
#pragma unroll
      for (int k = 0; k < 7; k++)
        acc += sI[nb][w + k] * w3[(o * 5 + nb) * 7 + k];
    }
    float s = g3[o] * rsqrtf(v3[o] + EPSV);
    float val = fmaxf((acc - m3[o]) * s + bb3[o], 0.f);
    h3[((size_t)bc * 32 + o) * 256 + w] = (_Float16)val;
  }
}

// ---------------------------------------------------------------------------
// Kernel 3: conv4 (32 -> 32, k(1,5)) via WMMA f16 + BN4 + ReLU + avgpool4.
// Grid: B*19 blocks, 256 threads. 16 mtile jobs; A shared by the 2 channel
// tiles; B is a contiguous v16h run from w4c. Each lane's 4 consecutive D
// rows form one pool bin (no cross-lane reduction).
// Out: X f16 [(b*19+c)][ch*64 + bin] (fc1 input layout).
// ---------------------------------------------------------------------------
__global__ __launch_bounds__(256) void k_conv4(
    const _Float16* __restrict__ h3, const _Float16* __restrict__ w4c,
    const float* __restrict__ b4, const float* __restrict__ g4,
    const float* __restrict__ bb4, const float* __restrict__ m4,
    const float* __restrict__ v4, _Float16* __restrict__ X) {
  __shared__ _Float16 sH[32][264];   // 260 used, w = -2..257
  int bc = blockIdx.x;
  int tid = threadIdx.x;
  for (int i = tid; i < 32 * 260; i += 256) {
    int ci = i / 260, j = i % 260;
    int w = j - 2;
    int wc = min(max(w, 0), 255);
    float val = (float)h3[((size_t)bc * 32 + ci) * 256 + wc];
    sH[ci][j] = (_Float16)((w == wc) ? val : 0.f);
  }
  __syncthreads();
  const int wave = tid >> 5, lane = tid & 31;
  const int mL = lane & 15, hi = lane >> 4;
  for (int mt = wave; mt < 16; mt += 8) {
    int w0 = mt * 16;
    v8f acc0 = {}, acc1 = {};
#pragma unroll
    for (int kc = 0; kc < 5; kc++) {
      v16h A;
#pragma unroll
      for (int s = 0; s < 16; s++) {
        int k = kc * 32 + a_slot_k(s, hi);   // always < 160
        int ci = k / 5, tp = k % 5;
        A[s] = sH[ci][w0 + mL + tp];
      }
      const v16h B0 = *(const v16h*)(w4c + ((size_t)mL * 160 + kc * 32 + hi * 16));
      const v16h B1 = *(const v16h*)(w4c + ((size_t)(16 + mL) * 160 + kc * 32 + hi * 16));
      acc0 = __builtin_amdgcn_wmma_f32_16x16x32_f16(false, A, false, B0,
                                                    (short)0, acc0, false, false);
      acc1 = __builtin_amdgcn_wmma_f32_16x16x32_f16(false, A, false, B1,
                                                    (short)0, acc1, false, false);
    }
#pragma unroll
    for (int nt = 0; nt < 2; nt++) {
      const v8f& acc = nt ? acc1 : acc0;
      int och = nt * 16 + mL;
      float s4 = g4[och] * rsqrtf(v4[och] + EPSV);
      float sh = bb4[och] + (b4[och] - m4[och]) * s4;
      float p0 = 0.f, p1 = 0.f;
#pragma unroll
      for (int v = 0; v < 8; v++) {
        float val = fmaxf(acc[v] * s4 + sh, 0.f);
        if (v < 4) p0 += val; else p1 += val;
      }
      int binBase = (w0 >> 2) + hi * 2;   // positions w0+8*hi.. -> 2 bins
      X[(size_t)bc * 2048 + och * 64 + binBase]     = (_Float16)(p0 * 0.25f);
      X[(size_t)bc * 2048 + och * 64 + binBase + 1] = (_Float16)(p1 * 0.25f);
    }
  }
}

// ---------------------------------------------------------------------------
// Kernel 4: fc1 (2048->32, WMMA, M=1216 rows) + ReLU, then fc2/fc3 scalar.
// Grid: 76 blocks (76*16 = 1216 rows), 32 threads (one wave).
// A = two aligned v8h (b128) runs per chunk; B = contiguous v16h run.
// ---------------------------------------------------------------------------
__global__ __launch_bounds__(32) void k_fc(
    const _Float16* __restrict__ X, const _Float16* __restrict__ fc1T,
    const float* __restrict__ fc1b, const float* __restrict__ fc2w,
    const float* __restrict__ fc2b, const float* __restrict__ fc3w,
    const float* __restrict__ fc3b, float* __restrict__ out) {
  __shared__ float sY[16][33];
  const int tile = blockIdx.x;
  const int lane = threadIdx.x;
  const int mL = lane & 15, hi = lane >> 4;
  const size_t row = (size_t)(tile * 16 + mL);
  v8f acc0 = {}, acc1 = {};
  for (int kc = 0; kc < 64; kc++) {
    const v8h a0 = *(const v8h*)(X + row * 2048 + kc * 32 + hi * 16);
    const v8h a1 = *(const v8h*)(X + row * 2048 + kc * 32 + hi * 16 + 8);
    v16h A;
#pragma unroll
    for (int s = 0; s < 8; s++) { A[s] = a0[s]; A[8 + s] = a1[s]; }
    const v16h B0 = *(const v16h*)(fc1T + (size_t)mL * 2048 + kc * 32 + hi * 16);
    const v16h B1 = *(const v16h*)(fc1T + (size_t)(16 + mL) * 2048 + kc * 32 + hi * 16);
    acc0 = __builtin_amdgcn_wmma_f32_16x16x32_f16(false, A, false, B0,
                                                  (short)0, acc0, false, false);
    acc1 = __builtin_amdgcn_wmma_f32_16x16x32_f16(false, A, false, B1,
                                                  (short)0, acc1, false, false);
  }
#pragma unroll
  for (int v = 0; v < 8; v++) {
    int m = v + 8 * hi;
    sY[m][mL]      = fmaxf(acc0[v] + fc1b[mL], 0.f);
    sY[m][16 + mL] = fmaxf(acc1[v] + fc1b[16 + mL], 0.f);
  }
  __syncthreads();
  if (lane < 16) {
    int r = tile * 16 + lane;          // r = b*19 + c, d_out flat (64,19)
    float y2[32];
    for (int o = 0; o < 32; o++) {
      float a = fc2b[o];
#pragma unroll
      for (int k = 0; k < 32; k++) a += fc2w[o * 32 + k] * sY[lane][k];
      y2[o] = a;
    }
    float lg = fc3b[0];
#pragma unroll
    for (int k = 0; k < 32; k++) lg += fc3w[k] * y2[k];
    out[r] = lg;
  }
}

// ---------------------------------------------------------------------------
// Host launcher. Flattened setup_inputs order:
//  0:x 1:w1 2:b1 3:bn1_g 4:bn1_b 5:w2 6:b2 7:bn2_g 8:bn2_b
//  9:w3 10:b3 11:bn3_g 12:bn3_b 13:w4 14:b4 15:bn4_g 16:bn4_b
//  17:fc1_w 18:fc1_b 19:fc2_w 20:fc2_b 21:fc3_w 22:fc3_b
//  23:m1 24:v1 25:m2 26:v2 27:m3 28:v3 29:m4 30:v4
// ---------------------------------------------------------------------------
extern "C" void kernel_launch(void* const* d_in, const int* in_sizes, int n_in,
                              void* d_out, int out_size, void* d_ws, size_t ws_size,
                              hipStream_t stream) {
  (void)in_sizes; (void)n_in; (void)out_size; (void)ws_size;
  const float* x    = (const float*)d_in[0];
  const float* w1   = (const float*)d_in[1];
  const float* b1   = (const float*)d_in[2];
  const float* g1   = (const float*)d_in[3];
  const float* bb1  = (const float*)d_in[4];
  const float* w2   = (const float*)d_in[5];
  const float* b2   = (const float*)d_in[6];
  const float* g2   = (const float*)d_in[7];
  const float* bb2  = (const float*)d_in[8];
  const float* w3   = (const float*)d_in[9];
  const float* b3   = (const float*)d_in[10];
  const float* g3   = (const float*)d_in[11];
  const float* bb3  = (const float*)d_in[12];
  const float* w4   = (const float*)d_in[13];
  const float* b4   = (const float*)d_in[14];
  const float* g4   = (const float*)d_in[15];
  const float* bb4  = (const float*)d_in[16];
  const float* fc1w = (const float*)d_in[17];
  const float* fc1b = (const float*)d_in[18];
  const float* fc2w = (const float*)d_in[19];
  const float* fc2b = (const float*)d_in[20];
  const float* fc3w = (const float*)d_in[21];
  const float* fc3b = (const float*)d_in[22];
  const float* m1   = (const float*)d_in[23];
  const float* v1   = (const float*)d_in[24];
  const float* m2   = (const float*)d_in[25];
  const float* v2   = (const float*)d_in[26];
  const float* m3   = (const float*)d_in[27];
  const float* v3   = (const float*)d_in[28];
  const float* m4   = (const float*)d_in[29];
  const float* v4   = (const float*)d_in[30];

  // Workspace carve-up (~31.3 MB, all offsets 256B-aligned)
  char* ws = (char*)d_ws;
  float*    pooled = (float*)(ws);                    // 1,556,480 f32 (6,225,920 B)
  _Float16* h3     = (_Float16*)(ws + 6225920);       // 9,961,472 f16 (19,922,944 B)
  _Float16* X      = (_Float16*)(ws + 26148864);      // 2,490,368 f16 ( 4,980,736 B)
  _Float16* w2g    = (_Float16*)(ws + 31129600);      // 15,360 f16 (30,720 B)
  _Float16* w4c    = (_Float16*)(ws + 31160320);      // 5,120 f16 (10,240 B)
  _Float16* fc1T   = (_Float16*)(ws + 31170560);      // 65,536 f16 (131,072 B)

  k_prep<<<256, 256, 0, stream>>>(w2, w4, fc1w, w2g, w4c, fc1T);

  dim3 gFront(64, 64);
  k_front<<<gFront, 256, 0, stream>>>(x, w1, b1, g1, bb1, m1, v1,
                                      w2g, b2, g2, bb2, m2, v2, pooled);

  k_conv3<<<1216, 128, 0, stream>>>(pooled, w3, b3, g3, bb3, m3, v3, h3);

  k_conv4<<<1216, 256, 0, stream>>>(h3, w4c, b4, g4, bb4, m4, v4, X);

  k_fc<<<76, 32, 0, stream>>>(X, fc1T, fc1b, fc2w, fc2b, fc3w, fc3b,
                              (float*)d_out);
}
